// GaussianUpsample_79937931313770
// MI455X (gfx1250) — compile-verified, matrix operand order checked by reading, not run on hardware
//
#include <hip/hip_runtime.h>
#include <math.h>
#include <stdint.h>

#define B_ 32
#define S_ 256
#define T_ 2048
#define D_ 256
#define BS 40   // B-tile LDS row stride in halves (80B: 16B-aligned, conflict-staggered)

typedef _Float16 half8  __attribute__((ext_vector_type(8)));
typedef _Float16 half16 __attribute__((ext_vector_type(16)));
typedef float    float8 __attribute__((ext_vector_type(8)));

// gfx1250 async global->LDS copy path (ASYNCcnt-tracked DMA), with portable fallback.
#if defined(__HIP_DEVICE_COMPILE__) && defined(__gfx1250__) && \
    __has_builtin(__builtin_amdgcn_global_load_async_to_lds_b128)
#define GU_ASYNC 1
typedef int v4i __attribute__((vector_size(4 * sizeof(int))));
typedef __attribute__((address_space(1))) v4i gv4i;   // global int4
typedef __attribute__((address_space(3))) v4i lv4i;   // LDS int4
__device__ __forceinline__ void gu_stage_row_async(_Float16* ldst, const _Float16* gsrc) {
  gv4i* g = (gv4i*)(uintptr_t)gsrc;            // AS1: full 64-bit VA
  lv4i* l = (lv4i*)(uintptr_t)ldst;            // AS3: low 32 bits = LDS offset
  __builtin_amdgcn_global_load_async_to_lds_b128(g, l, 0, 0);
  __builtin_amdgcn_global_load_async_to_lds_b128(g, l, 16, 0);
  __builtin_amdgcn_global_load_async_to_lds_b128(g, l, 32, 0);
  __builtin_amdgcn_global_load_async_to_lds_b128(g, l, 48, 0);
}
__device__ __forceinline__ void gu_wait_async0() {
#if __has_builtin(__builtin_amdgcn_s_wait_asynccnt)
  __builtin_amdgcn_s_wait_asynccnt(0);
#else
  asm volatile("s_wait_asynccnt 0x0" ::: "memory");
#endif
}
#else
#define GU_ASYNC 0
#endif

// ---------------------------------------------------------------------------
// Kernel 1: per-token prep: scan durations -> centers c; ir2 = 1/r^2.
// ---------------------------------------------------------------------------
__global__ void __launch_bounds__(S_) gu_prep_tokens(
    const float* __restrict__ dur, const float* __restrict__ rng,
    float* __restrict__ c_out, float* __restrict__ ir2_out) {
  __shared__ float sc[S_];
  const int b = blockIdx.x, s = threadIdx.x;
  const float d = dur[b * S_ + s];
  sc[s] = d;
  __syncthreads();
  for (int off = 1; off < S_; off <<= 1) {   // Hillis-Steele inclusive scan
    float v = (s >= off) ? sc[s - off] : 0.0f;
    __syncthreads();
    sc[s] += v;
    __syncthreads();
  }
  const float e = sc[s];
  c_out[b * S_ + s] = e - 0.5f * d;
  const float r = rng[b * S_ + s];
  ir2_out[b * S_ + s] = 1.0f / (r * r);
}

// ---------------------------------------------------------------------------
// Kernel 2: per-frame normalizer invw2[b,t] = 1/(sum_s exp(-ir2*(t-c)^2)+eps)
// ---------------------------------------------------------------------------
__global__ void __launch_bounds__(256) gu_prep_frames(
    const float* __restrict__ c_in, const float* __restrict__ ir2_in,
    float* __restrict__ invw2) {
  __shared__ float cs[S_], is[S_];
  const int b = blockIdx.y;
  const int tid = threadIdx.x;
  cs[tid] = c_in[b * S_ + tid];
  is[tid] = ir2_in[b * S_ + tid];
  __syncthreads();
  const int t = blockIdx.x * 256 + tid;
  const float ft = (float)t;
  float sum = 0.0f;
#pragma unroll 4
  for (int s = 0; s < S_; ++s) {
    const float dd = ft - cs[s];
    sum += __expf(-is[s] * dd * dd);
  }
  invw2[b * T_ + t] = 1.0f / (sum + 1e-20f);
}

// ---------------------------------------------------------------------------
// Kernel 3: sinusoidal PE table (T x D f32, 2MB, L2-resident, shared by all b)
// ---------------------------------------------------------------------------
__global__ void __launch_bounds__(256) gu_prep_pe(float* __restrict__ pe) {
  const int idx = blockIdx.x * 256 + threadIdx.x;
  const int t = idx >> 8;
  const int d = idx & (D_ - 1);
  const float inv = __expf(-9.21034037197618f / (float)D_ * (float)(d & ~1));
  const float ang = (float)t * inv;
  pe[idx] = (d & 1) ? __cosf(ang) : __sinf(ang);
}

// ---------------------------------------------------------------------------
// Kernel 4: enc (B,S,D) f32 -> encT (B,D,S) f16 via LDS transpose tile.
// ---------------------------------------------------------------------------
__global__ void __launch_bounds__(256) gu_prep_encT(
    const float* __restrict__ enc, _Float16* __restrict__ encT) {
  __shared__ float tile[64][65];
  const int b = blockIdx.z;
  const int s0 = blockIdx.y * 64, d0 = blockIdx.x * 64;
  const int tid = threadIdx.x;
  const int c0 = tid & 63, r0 = tid >> 6;
#pragma unroll
  for (int i = 0; i < 16; ++i) {
    const int sl = r0 + i * 4;
    tile[sl][c0] = enc[(size_t)b * S_ * D_ + (size_t)(s0 + sl) * D_ + d0 + c0];
  }
  __syncthreads();
#pragma unroll
  for (int i = 0; i < 16; ++i) {
    const int dd = r0 + i * 4;
    encT[(size_t)b * D_ * S_ + (size_t)(d0 + dd) * S_ + s0 + c0] =
        (_Float16)tile[c0][dd];
  }
}

// ---------------------------------------------------------------------------
// Kernel 5: fused weight-gen + GEMM + PE epilogue.
// Block = 256 threads (8 waves) -> 64(T) x 256(full D) tile of one batch.
// A fragments (normalized gaussian weights) live purely in registers.
// B tiles staged to LDS: async DMA double-buffer (gfx1250) or reg-pipelined.
// ---------------------------------------------------------------------------
__global__ void __launch_bounds__(256) gu_gemm(
    const _Float16* __restrict__ encT, const float* __restrict__ c_in,
    const float* __restrict__ ir2_in, const float* __restrict__ invw2,
    const float* __restrict__ pe, float* __restrict__ out) {
#if GU_ASYNC
  __shared__ _Float16 Blds[2][D_ * BS];   // double buffer, 2 x 20KB
#else
  __shared__ _Float16 Blds[1][D_ * BS];   // single buffer + register pipeline
#endif
  __shared__ float cs[S_], is[S_], iw[64];

  const int b  = blockIdx.y;
  const int t0 = blockIdx.x * 64;
  const int tid = threadIdx.x;

  cs[tid] = c_in[b * S_ + tid];
  is[tid] = ir2_in[b * S_ + tid];
  if (tid < 64) iw[tid] = invw2[b * T_ + t0 + tid];

  const int lane = tid & 31;
  const int wave = tid >> 5;
  const int wm = wave & 3;             // t sub-block (16 rows)
  const int wn = wave >> 2;            // d half (128 cols)
  const int r  = lane & 15;
  const int kb = (lane < 16) ? 0 : 8;  // fragment K-base per 16-bit ISA layout
  const int trow = wm * 16 + r;
  const float ft = (float)(t0 + trow);

  float8 acc[8] = {};
  const _Float16* encRow = encT + (size_t)b * D_ * S_ + (size_t)tid * S_;

#if GU_ASYNC
  gu_stage_row_async(&Blds[0][tid * BS], encRow);          // tile 0 in flight
#else
  const half8* gp0 = (const half8*)encRow;                 // prefetch tile 0 to regs
  half8 p0 = gp0[0], p1 = gp0[1], p2 = gp0[2], p3 = gp0[3];
#endif
  __syncthreads();                                         // cs/is/iw visible
  const float miw = iw[trow];

  for (int ik = 0; ik < 8; ++ik) {
    const int k0 = ik * 32;
    // --- A fragment in registers: 16 normalized gaussian weights ---
    half16 afrag = {};
#pragma unroll
    for (int j = 0; j < 8; ++j) {
      const int s = k0 + kb + j;
      const float dd = ft - cs[s];
      afrag[j] = (_Float16)(__expf(-is[s] * dd * dd) * miw);
      const int s2 = s + 16;
      const float dd2 = ft - cs[s2];
      afrag[8 + j] = (_Float16)(__expf(-is[s2] * dd2 * dd2) * miw);
    }

#if GU_ASYNC
    gu_wait_async0();                  // my DMA for tile ik done
    __syncthreads();                   // everyone's done; prior reads drained
    if (ik < 7)                        // overlap tile ik+1 DMA with WMMAs
      gu_stage_row_async(&Blds[(ik + 1) & 1][tid * BS], encRow + k0 + 32);
    const _Float16* curB = Blds[ik & 1];
#else
    __syncthreads();                   // all reads of previous tile drained
    {                                  // store prefetched regs, then prefetch next
      half8* dst = (half8*)&Blds[0][tid * BS];
      dst[0] = p0; dst[1] = p1; dst[2] = p2; dst[3] = p3;
    }
    if (ik < 7) {
      const half8* s = (const half8*)(encRow + k0 + 32);
      p0 = s[0]; p1 = s[1]; p2 = s[2]; p3 = s[3];
    }
    __syncthreads();                   // tile ik staged by all waves
    const _Float16* curB = Blds[0];
#endif

    // --- 8 WMMA tiles across this wave's 128 d-columns ---
#pragma unroll
    for (int i = 0; i < 8; ++i) {
      const _Float16* bp = &curB[(wn * 128 + 16 * i + r) * BS + kb];
      const half8 blo = *(const half8*)bp;
      const half8 bhi = *(const half8*)(bp + 16);
      const half16 bfrag = __builtin_shufflevector(
          blo, bhi, 0, 1, 2, 3, 4, 5, 6, 7, 8, 9, 10, 11, 12, 13, 14, 15);
      acc[i] = __builtin_amdgcn_wmma_f32_16x16x32_f16(
          false, afrag, false, bfrag, (short)0, acc[i], false, false);
    }
  }

  // --- epilogue: PE from L2-resident table; NT stores keep L2 for tables ---
  const int m = t0 + wm * 16 + ((lane < 16) ? 0 : 8);
  const int nBase = wn * 128 + (lane & 15);
#pragma unroll
  for (int i = 0; i < 8; ++i) {
    const int d = nBase + 16 * i;
#pragma unroll
    for (int v = 0; v < 8; ++v) {
      const int t = m + v;
      const float val = acc[i][v] + pe[t * D_ + d];
      __builtin_nontemporal_store(val, &out[(size_t)b * T_ * D_ + (size_t)t * D_ + d]);
    }
  }
}

// ---------------------------------------------------------------------------
extern "C" void kernel_launch(void* const* d_in, const int* in_sizes, int n_in,
                              void* d_out, int out_size, void* d_ws, size_t ws_size,
                              hipStream_t stream) {
  (void)in_sizes; (void)n_in; (void)out_size; (void)ws_size;
  const float* enc = (const float*)d_in[0];   // (B,S,D) f32
  const float* dur = (const float*)d_in[1];   // (B,S)   f32
  const float* rng = (const float*)d_in[2];   // (B,S,1) f32
  float* out = (float*)d_out;                 // (B,T,D) f32

  float* ws    = (float*)d_ws;
  float* c     = ws;                                  // B*S      (32KB)
  float* ir2   = c + B_ * S_;                         // B*S      (32KB)
  float* invw2 = ir2 + B_ * S_;                       // B*T      (256KB)
  float* pe    = invw2 + B_ * T_;                     // T*D      (2MB)
  _Float16* encT = (_Float16*)(pe + T_ * D_);         // B*D*S f16 (4MB)

  gu_prep_tokens<<<B_, S_, 0, stream>>>(dur, rng, c, ir2);
  gu_prep_frames<<<dim3(T_ / 256, B_), 256, 0, stream>>>(c, ir2, invw2);
  gu_prep_pe<<<(T_ * D_) / 256, 256, 0, stream>>>(pe);
  gu_prep_encT<<<dim3(D_ / 64, S_ / 64, B_), 256, 0, stream>>>(enc, encT);
  gu_gemm<<<dim3(T_ / 64, B_), 256, 0, stream>>>(encT, c, ir2, invw2, pe, out);
}
